// VanillaLSTM_old_50989851738134
// MI455X (gfx1250) — matrix operand order repaired and checked by reading, not compiled, not run on hardware
//
#include <hip/hip_runtime.h>
#include <hip/hip_bf16.h>
#include <cstdint>

// ---- problem sizes (fixed by the reference) ----
#define B_  64
#define S_  512
#define H_  1024
#define A_  16
#define G4H (4 * H_)
#define NBLK 64        // persistent-scan grid size (co-resident: << #WGPs)

typedef __attribute__((ext_vector_type(16))) __bf16 v16bf;
typedef __attribute__((ext_vector_type(8)))  float  v8f;

struct U16x16 { uint4 lo, hi; };   // 32B carrier for a v16bf fragment

__device__ __forceinline__ __bf16 f2bf(float f) { return (__bf16)f; }
__device__ __forceinline__ float  bf2f(__bf16 b) { return (float)b; }
__device__ __forceinline__ float  sigf(float x) { return 1.0f / (1.0f + __expf(-x)); }

// ---- CDNA5 async global->LDS copy (ASYNCcnt-tracked) ----
__device__ __forceinline__ void async_b128(void* ldsp, const void* gptr) {
  asm volatile("global_load_async_to_lds_b128 %0, %1, off"
               :: "v"((unsigned)(size_t)ldsp), "v"(gptr)
               : "memory");
}
__device__ __forceinline__ void async_wait0() { asm volatile("s_wait_asynccnt 0" ::: "memory"); }
__device__ __forceinline__ void async_wait2() { asm volatile("s_wait_asynccnt 2" ::: "memory"); }
__device__ __forceinline__ void async_wait3() { asm volatile("s_wait_asynccnt 3" ::: "memory"); }

// A fragment: ISA 16-bit A 16x32 layout (two contiguous 16B LDS reads)
__device__ __forceinline__ v16bf load_a_frag(const __bf16 (*At)[40], int ar, int hl) {
  U16x16 t;
  t.lo = *reinterpret_cast<const uint4*>(&At[ar][hl * 8]);
  t.hi = *reinterpret_cast<const uint4*>(&At[ar][16 + hl * 8]);
  return __builtin_bit_cast(v16bf, t);
}
// B fragment from column-major LDS tile BtT[n][k]
__device__ __forceinline__ v16bf load_b_frag(const __bf16 (*BtT)[40], int n, int hl) {
  U16x16 t;
  t.lo = *reinterpret_cast<const uint4*>(&BtT[n][hl * 8]);
  t.hi = *reinterpret_cast<const uint4*>(&BtT[n][16 + hl * 8]);
  return __builtin_bit_cast(v16bf, t);
}

// ---- init: zero dummy outputs, h0, and the grid-barrier counter ----
__global__ __launch_bounds__(256)
void init_k(float* __restrict__ out, __bf16* __restrict__ h, unsigned* __restrict__ bar) {
  int i = blockIdx.x * 256 + threadIdx.x;          // 0 .. B*H-1
  if (i < B_ * S_) {
    out[i] = 0.0f;
    out[(size_t)B_ * S_ * (1 + A_) + i] = 0.0f;
  }
  if (i == 0) *bar = 0u;
  h[i] = f2bf(0.0f);
}

// ---- one-time fp32 -> bf16 TRANSPOSED weight conversion: dst[N][K] <- src[K][N] ----
__global__ __launch_bounds__(256)
void cvt_transpose(const float* __restrict__ src, __bf16* __restrict__ dst,
                   int K, int N) {
  __shared__ float tile[32][33];
  const int tid = threadIdx.x;
  const int kb = blockIdx.y * 32, nb = blockIdx.x * 32;
  const int tx = tid & 31, ty = tid >> 5;
#pragma unroll
  for (int i = 0; i < 4; ++i)
    tile[ty + i * 8][tx] = src[(size_t)(kb + ty + i * 8) * N + nb + tx];
  __syncthreads();
#pragma unroll
  for (int i = 0; i < 4; ++i)
    dst[(size_t)(nb + ty + i * 8) * K + kb + tx] = f2bf(tile[tx][ty + i * 8]);
}

__global__ __launch_bounds__(256)
void cvt_transpose_wd(const float* __restrict__ src, __bf16* __restrict__ dst) {
  int idx = blockIdx.x * 256 + threadIdx.x;        // over [n][k]
  int n = idx >> 10, k = idx & (H_ - 1);
  dst[idx] = f2bf(src[(size_t)k * A_ + n]);
}

// ---- token + positional embedding -> bf16 GEMM operand ----
__global__ __launch_bounds__(256)
void embed_k(const int* __restrict__ x, const float* __restrict__ tok,
             const float* __restrict__ pos, __bf16* __restrict__ xe) {
  int idx = blockIdx.x * 256 + threadIdx.x;
  int bs = idx >> 10;
  int hh = idx & (H_ - 1);
  int s  = bs & (S_ - 1);
  int t  = x[bs];
  xe[idx] = f2bf(tok[(size_t)t * H_ + hh] + pos[(size_t)s * H_ + hh]);
}

// ---- 64x64-tile bf16 WMMA GEMM, double-buffered async pipeline ----
__global__ __launch_bounds__(256)
void gemm_tile64_bf16out(const __bf16* __restrict__ Am, const __bf16* __restrict__ BmT,
                         __bf16* __restrict__ C, int N, int K) {
  __shared__ __bf16 At[2][64][40];
  __shared__ __bf16 Bt[2][64][40];

  const int tid  = threadIdx.x;
  const int lane = tid & 31, wave = tid >> 5;
  const int hl   = lane >> 4, r16 = lane & 15;
  const int mw   = wave & 3,  nw2 = wave >> 2;
  const int rowBase = blockIdx.y * 64;
  const int colBase = blockIdx.x * 64;

  v8f acc0 = {}; v8f acc1 = {};

  const int rL = tid >> 2, sL = (tid & 3) * 8;
  const __bf16* aBase = Am  + (size_t)(rowBase + rL) * K + sL;
  const __bf16* bBase = BmT + (size_t)(colBase + rL) * K + sL;

  auto issue = [&](int buf, int k0) {
    async_b128(&At[buf][rL][sL], aBase + k0);
    async_b128(&Bt[buf][rL][sL], bBase + k0);
  };
  auto consume = [&](int b) {
    const v16bf af = load_a_frag(At[b], mw * 16 + r16, hl);
    const v16bf b0 = load_b_frag(Bt[b], nw2 * 32 + r16, hl);
    const v16bf b1 = load_b_frag(Bt[b], nw2 * 32 + r16 + 16, hl);
    acc0 = __builtin_amdgcn_wmma_f32_16x16x32_bf16(false, af, false, b0, (short)0, acc0, false, false);
    acc1 = __builtin_amdgcn_wmma_f32_16x16x32_bf16(false, af, false, b1, (short)0, acc1, false, false);
  };

  const int KT = K >> 5;
  issue(0, 0);
  int kc = 0;
  for (; kc < KT - 1; ++kc) {                      // steady state: no branches
    issue((kc + 1) & 1, (kc + 1) << 5);
    async_wait2();
    __syncthreads();
    consume(kc & 1);
    __syncthreads();
  }
  async_wait0();
  __syncthreads();
  consume(kc & 1);

#pragma unroll
  for (int r = 0; r < 8; ++r) {
    int row = rowBase + mw * 16 + hl * 8 + r;
    int c0  = colBase + nw2 * 32 + r16;
    C[(size_t)row * N + c0]      = f2bf(acc0[r]);
    C[(size_t)row * N + c0 + 16] = f2bf(acc1[r]);
  }
}

// ---- persistent LSTM scan: all 512 steps in ONE launch ----
// 64 co-resident blocks; block bx owns hidden columns n0..n0+15 of all four
// gates. Cell state c lives in REGISTERS for the whole scan. Steps are
// separated by a software grid barrier (agent-scope atomic + acquire/release).
__global__ __launch_bounds__(256)
void lstm_scan_persistent(__bf16* __restrict__ hA, __bf16* __restrict__ hB,
                          const __bf16* __restrict__ WhT,
                          const __bf16* __restrict__ xi, const float* __restrict__ bh,
                          __bf16* __restrict__ hs, unsigned* __restrict__ bar) {
  __shared__ __bf16 At[2][64][40];   // h_prev tiles
  __shared__ __bf16 Bt[2][64][40];   // WhT tiles: 64 logical cols (4 gates x 16)
  __shared__ float  gbuf[4][64][16];

  const int tid  = threadIdx.x;
  const int lane = tid & 31, wave = tid >> 5;
  const int hl   = lane >> 4, r16 = lane & 15;
  const int mw   = wave & 3,  nw2 = wave >> 2;
  const int n0   = blockIdx.x * 16;

  const int rL = tid >> 2, sL = (tid & 3) * 8;
  const int gateL = rL >> 4, ccL = rL & 15;
  const __bf16* bBase = WhT + (size_t)(gateL * H_ + n0 + ccL) * H_ + sL;   // loop-invariant

  // per-thread epilogue cell: idx = tid*4+q -> (batch row, col)
  float creg[4] = {0.f, 0.f, 0.f, 0.f};
  // per-thread bias, loop-invariant
  float bhv[4][4];
#pragma unroll
  for (int q = 0; q < 4; ++q) {
    int j = n0 + ((tid * 4 + q) & 15);
#pragma unroll
    for (int g = 0; g < 4; ++g) bhv[g][q] = bh[g * H_ + j];
  }

  const __bf16* hp = hA;
  __bf16*       hn = hB;

  for (int t = 0; t < S_; ++t) {
    const __bf16* aBase = hp + (size_t)rL * H_ + sL;
    v8f acc0 = {}; v8f acc1 = {};

    auto issue = [&](int buf, int k0) {
      async_b128(&At[buf][rL][sL], aBase + k0);
      async_b128(&Bt[buf][rL][sL], bBase + k0);
    };
    auto consume = [&](int b) {
      const v16bf af = load_a_frag(At[b], mw * 16 + r16, hl);
      const v16bf b0 = load_b_frag(Bt[b], nw2 * 32 + r16, hl);
      const v16bf b1 = load_b_frag(Bt[b], nw2 * 32 + r16 + 16, hl);
      acc0 = __builtin_amdgcn_wmma_f32_16x16x32_bf16(false, af, false, b0, (short)0, acc0, false, false);
      acc1 = __builtin_amdgcn_wmma_f32_16x16x32_bf16(false, af, false, b1, (short)0, acc1, false, false);
    };

    issue(0, 0);
    int kc = 0;
#pragma unroll 1
    for (; kc < (H_ >> 5) - 1; ++kc) {
      issue((kc + 1) & 1, (kc + 1) << 5);
      async_wait2();
      __syncthreads();
      consume(kc & 1);
      __syncthreads();
    }
    async_wait0();
    __syncthreads();
    consume(kc & 1);

    // publish gate accumulators: wave (mw,nw2) owns gates {2*nw2, 2*nw2+1}
#pragma unroll
    for (int r = 0; r < 8; ++r) {
      int row = mw * 16 + hl * 8 + r;
      gbuf[nw2 * 2][row][r16]     = acc0[r];
      gbuf[nw2 * 2 + 1][row][r16] = acc1[r];
    }
    __syncthreads();

    // epilogue: each thread finalizes its 4 (batch, hidden) cells
#pragma unroll
    for (int q = 0; q < 4; ++q) {
      int idx = tid * 4 + q;
      int row = idx >> 4;                  // batch
      int col = idx & 15;
      int j   = n0 + col;
      size_t xrow = ((size_t)row * S_ + t) * (size_t)G4H;
      float zi = gbuf[0][row][col] + bf2f(xi[xrow + j])          + bhv[0][q];
      float zf = gbuf[1][row][col] + bf2f(xi[xrow + H_ + j])     + bhv[1][q];
      float zg = gbuf[2][row][col] + bf2f(xi[xrow + 2 * H_ + j]) + bhv[2][q];
      float zo = gbuf[3][row][col] + bf2f(xi[xrow + 3 * H_ + j]) + bhv[3][q];
      float cn = sigf(zf) * creg[q] + sigf(zi) * tanhf(zg);
      float hv = sigf(zo) * tanhf(cn);
      creg[q] = cn;
      __bf16 hb = f2bf(hv);
      hn[(size_t)row * H_ + j] = hb;
      hs[((size_t)row * S_ + t) * H_ + j] = hb;
    }

    // ---- software grid barrier between steps ----
    __threadfence();                       // release this block's hn writes
    __syncthreads();
    if (tid == 0) {
      __hip_atomic_fetch_add(bar, 1u, __ATOMIC_ACQ_REL, __HIP_MEMORY_SCOPE_AGENT);
      const unsigned target = (unsigned)(t + 1) * NBLK;
      while (__hip_atomic_load(bar, __ATOMIC_ACQUIRE, __HIP_MEMORY_SCOPE_AGENT) < target)
        __builtin_amdgcn_s_sleep(1);
    }
    __syncthreads();

    __bf16* tmp = hn; hn = (__bf16*)hp; hp = tmp;
  }
}

// ---- head GEMM: logits[32768,16] = hs @ Wd + bd (WdT is [16][1024]) ----
__global__ __launch_bounds__(128)
void logits_tile(const __bf16* __restrict__ Hs, const __bf16* __restrict__ WdT,
                 const float* __restrict__ bd, float* __restrict__ out) {
  __shared__ __bf16 At[2][64][40];
  __shared__ __bf16 Bt[2][16][40];
  const int tid  = threadIdx.x;                    // 128 threads = 4 waves
  const int lane = tid & 31, wave = tid >> 5;
  const int hl   = lane >> 4, r16 = lane & 15;
  const int rowBase = blockIdx.x * 64;
  v8f acc = {};

  // wave-uniform: every thread issues exactly 3 asyncs per chunk
  const int s0 = tid * 2,     a0r = s0 >> 2, a0s = (s0 & 3) * 8;
  const int s1 = tid * 2 + 1, a1r = s1 >> 2, a1s = (s1 & 3) * 8;
  const int bseg = tid & 63,  bn = bseg >> 2, bs = (bseg & 3) * 8;

  auto issue = [&](int buf, int k0) {
    async_b128(&At[buf][a0r][a0s], Hs + (size_t)(rowBase + a0r) * H_ + k0 + a0s);
    async_b128(&At[buf][a1r][a1s], Hs + (size_t)(rowBase + a1r) * H_ + k0 + a1s);
    async_b128(&Bt[buf][bn][bs],   WdT + (size_t)bn * H_ + k0 + bs);
  };
  auto consume = [&](int b) {
    const v16bf af  = load_a_frag(At[b], wave * 16 + r16, hl);
    const v16bf bf_ = load_b_frag(Bt[b], r16, hl);
    acc = __builtin_amdgcn_wmma_f32_16x16x32_bf16(false, af, false, bf_, (short)0, acc, false, false);
  };

  issue(0, 0);
  int kc = 0;
  for (; kc < (H_ >> 5) - 1; ++kc) {
    issue((kc + 1) & 1, (kc + 1) << 5);
    async_wait3();
    __syncthreads();
    consume(kc & 1);
    __syncthreads();
  }
  async_wait0();
  __syncthreads();
  consume(kc & 1);

#pragma unroll
  for (int r = 0; r < 8; ++r) {
    int row = rowBase + wave * 16 + hl * 8 + r;
    out[(size_t)B_ * S_ + (size_t)row * A_ + r16] = acc[r] + bd[r16];
  }
}

extern "C" void kernel_launch(void* const* d_in, const int* in_sizes, int n_in,
                              void* d_out, int out_size, void* d_ws, size_t ws_size,
                              hipStream_t stream) {
  (void)in_sizes; (void)n_in; (void)out_size; (void)ws_size;
  const int*   x   = (const int*)  d_in[0];
  // d_in[1..3] (true_actions, true_states, use_forcing) unused by reference math
  const float* tok = (const float*)d_in[4];
  const float* pos = (const float*)d_in[5];
  const float* Wi  = (const float*)d_in[6];
  const float* Wh  = (const float*)d_in[7];
  const float* bh  = (const float*)d_in[8];
  const float* Wd  = (const float*)d_in[9];
  const float* bd  = (const float*)d_in[10];
  float* out = (float*)d_out;

  char* ws = (char*)d_ws;
  size_t off = 0;
  auto carve = [&](size_t bytes) {
    void* p = ws + off;
    off = (off + bytes + 255) & ~(size_t)255;
    return p;
  };
  __bf16*   WiT  = (__bf16*)carve((size_t)H_ * G4H * 2);        //   8 MB, [4096][1024]
  __bf16*   WhT  = (__bf16*)carve((size_t)H_ * G4H * 2);        //   8 MB (L2-resident)
  __bf16*   WdT  = (__bf16*)carve((size_t)H_ * A_ * 2);         //  [16][1024]
  __bf16*   xe_b = (__bf16*)carve((size_t)B_ * S_ * H_ * 2);    //  64 MB
  __bf16*   xi_b = (__bf16*)carve((size_t)B_ * S_ * G4H * 2);   // 256 MB
  __bf16*   hs_b = (__bf16*)carve((size_t)B_ * S_ * H_ * 2);    //  64 MB
  __bf16*   hA   = (__bf16*)carve((size_t)B_ * H_ * 2);
  __bf16*   hB   = (__bf16*)carve((size_t)B_ * H_ * 2);
  unsigned* bar  = (unsigned*)carve(256);

  // 1) state + dummy-output + barrier init (every call)
  init_k<<<(B_ * H_) / 256, 256, 0, stream>>>(out, hA, bar);

  // 2) one-time transposed bf16 weights: W_T[N][K]
  cvt_transpose<<<dim3(G4H / 32, H_ / 32), 256, 0, stream>>>(Wi, WiT, H_, G4H);
  cvt_transpose<<<dim3(G4H / 32, H_ / 32), 256, 0, stream>>>(Wh, WhT, H_, G4H);
  cvt_transpose_wd<<<(A_ * H_) / 256, 256, 0, stream>>>(Wd, WdT);

  // 3) embedding gather + pos add
  embed_k<<<(B_ * S_ * H_) / 256, 256, 0, stream>>>(x, tok, pos, xe_b);

  // 4) hoisted input GEMM: xi = xe @ Wi
  gemm_tile64_bf16out<<<dim3(G4H / 64, (B_ * S_) / 64), 256, 0, stream>>>(
      xe_b, WiT, xi_b, G4H, H_);

  // 5) persistent scan: all 512 steps, ONE launch, c in registers
  lstm_scan_persistent<<<NBLK, 256, 0, stream>>>(hA, hB, WhT, xi_b, bh, hs_b, bar);

  // 6) head: logits = hs @ Wd + bd
  logits_tile<<<(B_ * S_) / 64, 128, 0, stream>>>(hs_b, WdT, bd, out);
}
